// FusedLinearCrossEntropyLoss_6347961663553
// MI455X (gfx1250) — matrix operand order repaired and checked by reading, not compiled, not run on hardware
//
#include <hip/hip_runtime.h>
#include <math.h>

// Problem constants (B=4, T=512, NQ=2, D=1024, V=32000)
#define TOK   4096      // B*T*NQ tokens
#define DIM   1024
#define VOCAB 32000
#define MBLK  32        // token rows per workgroup
#define PARTS 4         // vocab partitions (one per wave pair)
#define COLS_PER_PART (VOCAB / PARTS)   // 8000
#define LDA   1032      // padded bf16 row stride in LDS (2064B: conflict-free b128 reads)
#define WROW  2048      // combined hi|lo row stride in split-W workspace (bf16 elems)

typedef __bf16 bf16;
typedef __attribute__((ext_vector_type(16))) __bf16       bf16x16;
typedef __attribute__((ext_vector_type(4)))  __bf16       bf16x4;
typedef __attribute__((ext_vector_type(8)))  float        f32x8;
typedef __attribute__((ext_vector_type(4)))  unsigned int u32x4;

union FragB { bf16x16 v; u32x4 q[2]; };
union AccU  { f32x8 v; float f[8]; };

// ---------------------------------------------------------------------------
// Kernel 1: split W (fp32) into interleaved bf16 hi|lo rows:
//   Wc[v*2048 + d]        = bf16(W[v][d])            (hi)
//   Wc[v*2048 + 1024 + d] = bf16(W[v][d] - hi)       (lo)
// so the GEMM needs ONE base pointer per vocab column stream.
// ---------------------------------------------------------------------------
__global__ void convert_w(const float* __restrict__ W, bf16* __restrict__ Wc, int n4) {
    int i = blockIdx.x * blockDim.x + threadIdx.x;
    const int stride = gridDim.x * blockDim.x;
    const float4* W4 = (const float4*)W;
    for (; i < n4; i += stride) {
        float4 w = W4[i];
        const int e = i * 4;               // element index into V*D
        const int v = e >> 10;             // vocab row
        const int d = e & 1023;            // dim offset (multiple of 4)
        bf16x4 h, l;
        float x;
        x = w.x; h[0] = (bf16)x; l[0] = (bf16)(x - (float)h[0]);
        x = w.y; h[1] = (bf16)x; l[1] = (bf16)(x - (float)h[1]);
        x = w.z; h[2] = (bf16)x; l[2] = (bf16)(x - (float)h[2]);
        x = w.w; h[3] = (bf16)x; l[3] = (bf16)(x - (float)h[3]);
        bf16* row = Wc + (size_t)v * WROW + d;
        *(bf16x4*)(row)        = h;
        *(bf16x4*)(row + 1024) = l;
    }
}

// ---------------------------------------------------------------------------
// Kernel 2: exact fp32 target logit: dot(hidden[n], W[target[n]]) + b[target]
// one 32-lane wave per token, 8 tokens per 256-thread block
// ---------------------------------------------------------------------------
__global__ void target_logit(const float* __restrict__ hidden,
                             const float* __restrict__ W,
                             const float* __restrict__ bias,
                             const int*   __restrict__ targets,
                             float* __restrict__ tlog) {
    const int lane = threadIdx.x & 31;
    const int tok  = blockIdx.x * (blockDim.x >> 5) + (threadIdx.x >> 5);
    if (tok >= TOK) return;
    const int tgt = targets[tok];
    const float4* h4 = (const float4*)(hidden + (size_t)tok * DIM);
    const float4* w4 = (const float4*)(W + (size_t)tgt * DIM);
    float acc = 0.f;
#pragma unroll
    for (int i = 0; i < DIM / 4 / 32; ++i) {
        float4 a = h4[i * 32 + lane];
        float4 b = w4[i * 32 + lane];
        acc += a.x * b.x + a.y * b.y + a.z * b.z + a.w * b.w;
    }
#pragma unroll
    for (int off = 16; off > 0; off >>= 1) acc += __shfl_xor(acc, off, 32);
    if (lane == 0) tlog[tok] = acc + bias[tgt];
}

// ---------------------------------------------------------------------------
// Kernel 3: WMMA GEMM (bf16x3 split ~= fp32) + online softmax statistics
// 256 threads = 8 waves: wave = (m_tile in {0,1}) x (vocab part in {0..3})
//
// Fragment layouts as loaded here (C per CDNA5 ISA):
//   A 16x32 bf16: lane half 0 -> K 0..15 contiguous;
//                 lane half 1 -> K 16..31. lane&15 = row M.
//   B 32x16 bf16: same K order as A here: half 0 -> K 0..15; half 1 -> K 16..31.
//                 lane&15 = col N.  (one contiguous 32B span per lane)
//   C 16x16 f32:  VGPR j, half h -> row M = 8h+j, col N = lane&15.
// ---------------------------------------------------------------------------
__global__ __launch_bounds__(256)
void gemm_softmax(const float* __restrict__ hidden,
                  const bf16*  __restrict__ Wc,
                  const float* __restrict__ bias,
                  float* __restrict__ smax, float* __restrict__ ssum) {
    __shared__ __align__(16) bf16 Ahi[MBLK * LDA];
    __shared__ __align__(16) bf16 Alo[MBLK * LDA];

    const int tid     = threadIdx.x;
    const int rowbase = blockIdx.x * MBLK;

    // Cooperative load + hi/lo split of the A block (32 rows x 1024) into LDS
    {
        const float4* h4 = (const float4*)(hidden + (size_t)rowbase * DIM);
        for (int i = tid; i < MBLK * DIM / 4; i += 256) {
            const int r  = i >> 8;        // DIM/4 = 256 float4 per row
            const int c4 = i & 255;
            float4 w = h4[(size_t)r * (DIM / 4) + c4];
            const int base = r * LDA + c4 * 4;
            float x; bf16 h;
            x = w.x; h = (bf16)x; Ahi[base + 0] = h; Alo[base + 0] = (bf16)(x - (float)h);
            x = w.y; h = (bf16)x; Ahi[base + 1] = h; Alo[base + 1] = (bf16)(x - (float)h);
            x = w.z; h = (bf16)x; Ahi[base + 2] = h; Alo[base + 2] = (bf16)(x - (float)h);
            x = w.w; h = (bf16)x; Ahi[base + 3] = h; Alo[base + 3] = (bf16)(x - (float)h);
        }
    }
    __syncthreads();

    const int lane = tid & 31;
    const int wid  = tid >> 5;
    const int mt   = wid & 1;        // which 16-row tile
    const int part = wid >> 1;       // vocab partition
    const int half = lane >> 4;      // lane half
    const int l16  = lane & 15;

    const int abase = (mt * 16 + l16) * LDA;
    const int boff  = half * 16;     // B: half selects K 0..15 vs 16..31

    float rm[8], rs[8];
#pragma unroll
    for (int j = 0; j < 8; ++j) { rm[j] = -__builtin_inff(); rs[j] = 0.f; }

    const int c0 = part * COLS_PER_PART;
    const int c1 = c0 + COLS_PER_PART;

    for (int n0 = c0; n0 < c1; n0 += 32) {
        AccU acc0, acc1;
#pragma unroll
        for (int j = 0; j < 8; ++j) { acc0.f[j] = 0.f; acc1.f[j] = 0.f; }

        // One base pointer per 16-column tile; hi/lo and K chunks are
        // compile-time immediate offsets on the b128 loads.
        const bf16* b0p = Wc + (size_t)(n0      + l16) * WROW + boff;
        const bf16* b1p = Wc + (size_t)(n0 + 16 + l16) * WROW + boff;

#pragma unroll 2
        for (int k0 = 0; k0 < DIM; k0 += 32) {
            const int ao = abase + k0 + half * 16;
            FragB ah, al;
            ah.q[0] = *(const u32x4*)(Ahi + ao);
            ah.q[1] = *(const u32x4*)(Ahi + ao + 8);
            al.q[0] = *(const u32x4*)(Alo + ao);
            al.q[1] = *(const u32x4*)(Alo + ao + 8);

            FragB bh0, bl0, bh1, bl1;
            bh0.q[0] = *(const u32x4*)(b0p + k0);
            bh0.q[1] = *(const u32x4*)(b0p + k0 + 8);
            bl0.q[0] = *(const u32x4*)(b0p + k0 + 1024);
            bl0.q[1] = *(const u32x4*)(b0p + k0 + 1032);
            bh1.q[0] = *(const u32x4*)(b1p + k0);
            bh1.q[1] = *(const u32x4*)(b1p + k0 + 8);
            bl1.q[0] = *(const u32x4*)(b1p + k0 + 1024);
            bl1.q[1] = *(const u32x4*)(b1p + k0 + 1032);

            // D = Ahi*Bhi + Ahi*Blo + Alo*Bhi  (~fp32 accuracy, f32 accumulate)
            acc0.v = __builtin_amdgcn_wmma_f32_16x16x32_bf16(false, ah.v, false, bh0.v, (short)0, acc0.v, false, false);
            acc0.v = __builtin_amdgcn_wmma_f32_16x16x32_bf16(false, ah.v, false, bl0.v, (short)0, acc0.v, false, false);
            acc0.v = __builtin_amdgcn_wmma_f32_16x16x32_bf16(false, al.v, false, bh0.v, (short)0, acc0.v, false, false);
            acc1.v = __builtin_amdgcn_wmma_f32_16x16x32_bf16(false, ah.v, false, bh1.v, (short)0, acc1.v, false, false);
            acc1.v = __builtin_amdgcn_wmma_f32_16x16x32_bf16(false, ah.v, false, bl1.v, (short)0, acc1.v, false, false);
            acc1.v = __builtin_amdgcn_wmma_f32_16x16x32_bf16(false, al.v, false, bh1.v, (short)0, acc1.v, false, false);
        }

        // Online softmax update; each row lives in a 16-lane group.
        const float b0 = bias[n0 + l16];
        const float b1 = bias[n0 + 16 + l16];
#pragma unroll
        for (int j = 0; j < 8; ++j) {
            float v = acc0.f[j] + b0;
            float tm = v;
            tm = fmaxf(tm, __shfl_xor(tm, 1, 32));
            tm = fmaxf(tm, __shfl_xor(tm, 2, 32));
            tm = fmaxf(tm, __shfl_xor(tm, 4, 32));
            tm = fmaxf(tm, __shfl_xor(tm, 8, 32));
            float nm = fmaxf(rm[j], tm);
            float p = __expf(v - nm);
            p += __shfl_xor(p, 1, 32);
            p += __shfl_xor(p, 2, 32);
            p += __shfl_xor(p, 4, 32);
            p += __shfl_xor(p, 8, 32);
            rs[j] = rs[j] * __expf(rm[j] - nm) + p;
            rm[j] = nm;

            v = acc1.f[j] + b1;
            tm = v;
            tm = fmaxf(tm, __shfl_xor(tm, 1, 32));
            tm = fmaxf(tm, __shfl_xor(tm, 2, 32));
            tm = fmaxf(tm, __shfl_xor(tm, 4, 32));
            tm = fmaxf(tm, __shfl_xor(tm, 8, 32));
            nm = fmaxf(rm[j], tm);
            p = __expf(v - nm);
            p += __shfl_xor(p, 1, 32);
            p += __shfl_xor(p, 2, 32);
            p += __shfl_xor(p, 4, 32);
            p += __shfl_xor(p, 8, 32);
            rs[j] = rs[j] * __expf(rm[j] - nm) + p;
            rm[j] = nm;
        }
    }

    // Every lane of a 16-group holds identical row stats; lanes 0 and 16 write.
    if (l16 == 0) {
        const int mrow = half * 8;
#pragma unroll
        for (int j = 0; j < 8; ++j) {
            const int g = rowbase + mt * 16 + mrow + j;
            smax[g * PARTS + part] = rm[j];
            ssum[g * PARTS + part] = rs[j];
        }
    }
}

// ---------------------------------------------------------------------------
// Kernel 4: merge partials, mask, deterministic tree-reduce to scalar loss
// ---------------------------------------------------------------------------
__global__ void combine_loss(const float* __restrict__ smax,
                             const float* __restrict__ ssum,
                             const float* __restrict__ tlog,
                             const int*   __restrict__ targets,
                             const int*   __restrict__ prefix_len,
                             float* __restrict__ out) {
    __shared__ float sl[256], sc[256];
    const int tid = threadIdx.x;
    float accL = 0.f, accC = 0.f;
    for (int n = tid; n < TOK; n += 256) {
        float m = -__builtin_inff();
#pragma unroll
        for (int p = 0; p < PARTS; ++p) m = fmaxf(m, smax[n * PARTS + p]);
        float s = 0.f;
#pragma unroll
        for (int p = 0; p < PARTS; ++p) s += ssum[n * PARTS + p] * __expf(smax[n * PARTS + p] - m);
        const float nll = -(tlog[n] - m - logf(s));
        const int tgt  = targets[n];
        const int b    = n >> 10;            // n / (T*NQ) = n / 1024
        const int tpos = (n >> 1) & 511;     // (n / NQ) % T
        if (tgt != 0 && tpos >= prefix_len[b]) { accL += nll; accC += 1.f; }
    }
    sl[tid] = accL; sc[tid] = accC;
    __syncthreads();
    for (int o = 128; o > 0; o >>= 1) {
        if (tid < o) { sl[tid] += sl[tid + o]; sc[tid] += sc[tid + o]; }
        __syncthreads();
    }
    if (tid == 0) out[0] = sl[0] / fmaxf(sc[0], 1.f);
}

// ---------------------------------------------------------------------------
extern "C" void kernel_launch(void* const* d_in, const int* in_sizes, int n_in,
                              void* d_out, int out_size, void* d_ws, size_t ws_size,
                              hipStream_t stream) {
    (void)in_sizes; (void)n_in; (void)out_size; (void)ws_size;
    const float* hidden  = (const float*)d_in[0];
    const int*   targets = (const int*)d_in[1];
    const int*   prefix  = (const int*)d_in[2];
    const float* W       = (const float*)d_in[3];
    const float* bias    = (const float*)d_in[4];

    char* ws = (char*)d_ws;
    size_t off = 0;
    bf16* Wc = (bf16*)(ws + off); off += (size_t)VOCAB * WROW * sizeof(bf16);
    float* smax = (float*)(ws + off); off += (size_t)TOK * PARTS * sizeof(float);
    float* ssum = (float*)(ws + off); off += (size_t)TOK * PARTS * sizeof(float);
    float* tlog = (float*)(ws + off); off += (size_t)TOK * sizeof(float);

    convert_w<<<4096, 256, 0, stream>>>(W, Wc, VOCAB * DIM / 4);
    target_logit<<<TOK / 8, 256, 0, stream>>>(hidden, W, bias, targets, tlog);
    gemm_softmax<<<TOK / MBLK, 256, 0, stream>>>(hidden, Wc, bias, smax, ssum);
    combine_loss<<<1, 256, 0, stream>>>(smax, ssum, tlog, targets, prefix, (float*)d_out);
}
